// LatentGuidedAttention_20220706030256
// MI455X (gfx1250) — compile-verified, matrix operand order checked by reading, not run on hardware
//
#include <hip/hip_runtime.h>
#include <hip/hip_bf16.h>

// ---------------------------------------------------------------------------
// LatentGuidedAttention for MI455X (gfx1250, wave32, WMMA 16x16x32 f16,
// Tensor Data Mover for K tiles in flash attention)
// ---------------------------------------------------------------------------
#define X_DIM   512
#define Z_DIM   16
#define HEADS   8
#define DIM_HEAD 64
#define INNER   512
#define BATCH   4
#define N_SEQ   2048
#define M_ROWS  (BATCH * N_SEQ)          // 8192
#define EPS     1e-5f

typedef _Float16 v16h __attribute__((ext_vector_type(16)));
typedef _Float16 v8h  __attribute__((ext_vector_type(8)));
typedef float    v8f  __attribute__((ext_vector_type(8)));
typedef unsigned int u32x4 __attribute__((ext_vector_type(4)));
typedef int          i32x4 __attribute__((ext_vector_type(4)));
typedef int          i32x8 __attribute__((ext_vector_type(8)));

#if defined(__HIP_DEVICE_COMPILE__) && __has_builtin(__builtin_amdgcn_tensor_load_to_lds)
#define USE_TDM 1
#else
#define USE_TDM 0
#endif

__device__ __forceinline__ v8f vzero8() {
    v8f z = {0.f,0.f,0.f,0.f,0.f,0.f,0.f,0.f};
    return z;
}

// Load a 16-half WMMA fragment: two contiguous 16B chunks at p and p+16 halves.
// Matches CDNA5 16-bit A/B layout: lane<16 holds K={0..7,16..23}, lane>=16 holds
// K={8..15,24..31} when the caller adds koff = (lane>=16)*8 to p.
__device__ __forceinline__ v16h load_frag16(const _Float16* p) {
    v8h lo = *(const v8h*)(p);
    v8h hi = *(const v8h*)(p + 16);
    v16h r;
#pragma unroll
    for (int i = 0; i < 8; ++i) { r[i] = lo[i]; r[i + 8] = hi[i]; }
    return r;
}

__device__ __forceinline__ v8f wmma_f16(v16h a, v16h b, v8f c) {
    return __builtin_amdgcn_wmma_f32_16x16x32_f16(false, a, false, b,
                                                  (short)0, c, false, false);
}

#if USE_TDM
// Tensor Data Mover: load a 2D tile of 32 rows x 64 halves (2B elements),
// row stride 512 elements, from global `gaddr` into LDS at byte offset
// `lds_addr`. D# built per CDNA5 ISA 8.3/8.4 (group0/group1 bitfields).
__device__ __forceinline__ void tdm_load_k_tile(unsigned lds_addr,
                                                unsigned long long gaddr) {
    u32x4 g0;
    g0[0] = 1u;                                          // count=1, user D#
    g0[1] = lds_addr;                                    // lds_addr [63:32]
    g0[2] = (unsigned)(gaddr & 0xffffffffull);           // global_addr[31:0]
    g0[3] = (unsigned)((gaddr >> 32) & 0x1ffffffull)     // global_addr[56:32]
          | (2u << 30);                                  // type=2 ("image")
    i32x8 g1;
    g1[0] = 0x00010000;                  // workgroup_mask=0, data_size=1 (2B)
    g1[1] = (int)(64u << 16);            // tensor_dim0[15:0]=64 in bits[63:48]
    g1[2] = (int)(32u << 16);            // tensor_dim1[15:0]=32 in bits[95:80]
    g1[3] = (int)(64u << 16);            // tile_dim0=64 in bits[127:112]
    g1[4] = 32;                          // tile_dim1=32 in bits[143:128]
    g1[5] = 512;                         // tensor_dim0_stride=512 (elements)
    g1[6] = 0;
    g1[7] = 0;
    i32x4 z4 = {0, 0, 0, 0};
#if defined(__clang_major__) && __clang_major__ >= 23
    i32x8 z8 = {0, 0, 0, 0, 0, 0, 0, 0};
    __builtin_amdgcn_tensor_load_to_lds(g0, g1, z4, z4, z8, 0);
#else
    __builtin_amdgcn_tensor_load_to_lds(g0, g1, z4, z4, 0);
#endif
}
#endif

// ---------------------------------------------------------------------------
// Kernel 1: f32 weight [K,Nc] -> f16 transposed [Nc,K] (fragment-friendly)
// ---------------------------------------------------------------------------
__global__ void __launch_bounds__(256)
cvt_transpose_kernel(const float* __restrict__ W, _Float16* __restrict__ Wt,
                     int K, int Nc) {
    int idx = blockIdx.x * 256 + threadIdx.x;     // coalesced read of W
    int kk = idx / Nc;
    int n  = idx % Nc;
    Wt[(size_t)n * K + kk] = (_Float16)W[idx];
}

// ---------------------------------------------------------------------------
// Kernel 2: z path: LN(z) @ Wz -> SiLU -> z1z2 [B,1024] f32
// ---------------------------------------------------------------------------
__global__ void __launch_bounds__(256)
zpath_kernel(const float* __restrict__ z, const float* __restrict__ g,
             const float* __restrict__ bb, const float* __restrict__ Wz,
             float* __restrict__ z1z2) {
    int b = blockIdx.x;
    const float* zr = z + b * Z_DIM;
    float zn[Z_DIM];
    float mu = 0.f;
#pragma unroll
    for (int i = 0; i < Z_DIM; ++i) mu += zr[i];
    mu *= (1.f / Z_DIM);
    float var = 0.f;
#pragma unroll
    for (int i = 0; i < Z_DIM; ++i) { float d = zr[i] - mu; var += d * d; }
    var *= (1.f / Z_DIM);
    float rs = rsqrtf(var + EPS);
#pragma unroll
    for (int i = 0; i < Z_DIM; ++i) zn[i] = (zr[i] - mu) * rs * g[i] + bb[i];

    for (int c = threadIdx.x; c < 2 * X_DIM; c += 256) {
        float a = 0.f;
#pragma unroll
        for (int i = 0; i < Z_DIM; ++i) a += zn[i] * Wz[i * (2 * X_DIM) + c];
        float sg = 1.f / (1.f + __expf(-a));
        z1z2[b * (2 * X_DIM) + c] = a * sg;        // SiLU
    }
}

// ---------------------------------------------------------------------------
// Kernel 3: x_mod = LN(x)*z1 + z2  -> f16 [M_ROWS, 512]
// one block (128 threads) per row
// ---------------------------------------------------------------------------
__global__ void __launch_bounds__(128)
xmod_kernel(const float* __restrict__ x, const float* __restrict__ g,
            const float* __restrict__ bb, const float* __restrict__ z1z2,
            _Float16* __restrict__ out) {
    int row = blockIdx.x;                // 0..8191
    int b = row >> 11;                   // row / 2048
    const float* xr = x + (size_t)row * X_DIM;
    int lane = threadIdx.x & 31;
    int wave = threadIdx.x >> 5;
    int i0 = threadIdx.x * 4;

    float4 xv = *(const float4*)(xr + i0);
    float s  = xv.x + xv.y + xv.z + xv.w;
    float sq = xv.x*xv.x + xv.y*xv.y + xv.z*xv.z + xv.w*xv.w;
#pragma unroll
    for (int off = 16; off >= 1; off >>= 1) {
        s  += __shfl_xor(s,  off, 32);
        sq += __shfl_xor(sq, off, 32);
    }
    __shared__ float ss[4], ssq[4];
    if (lane == 0) { ss[wave] = s; ssq[wave] = sq; }
    __syncthreads();
    s  = ss[0] + ss[1] + ss[2] + ss[3];
    sq = ssq[0] + ssq[1] + ssq[2] + ssq[3];
    float mu  = s * (1.f / X_DIM);
    float var = sq * (1.f / X_DIM) - mu * mu;
    float rs = rsqrtf(var + EPS);

    const float* z1 = z1z2 + b * (2 * X_DIM);
    const float* z2 = z1 + X_DIM;
    _Float16* op = out + (size_t)row * X_DIM + i0;
    float xs[4] = {xv.x, xv.y, xv.z, xv.w};
#pragma unroll
    for (int e = 0; e < 4; ++e) {
        int idx = i0 + e;
        float v = ((xs[e] - mu) * rs * g[idx] + bb[idx]) * z1[idx] + z2[idx];
        op[e] = (_Float16)v;
    }
}

// ---------------------------------------------------------------------------
// Kernel 4: WMMA GEMM  C[M,Nc] = A[M,K] @ Bt[Nc,K]^T  (Bt stored N-major)
// 128x128 block tile, 8 waves (4x2), 32x64 wave tile, k-step 32
// outF != null -> f32 output with bias; else f16 output
// ---------------------------------------------------------------------------
__global__ void __launch_bounds__(256)
gemm_wmma_kernel(const _Float16* __restrict__ A, const _Float16* __restrict__ Bt,
                 _Float16* __restrict__ outH, float* __restrict__ outF,
                 const float* __restrict__ bias, int M, int Nc, int K) {
    const int tid  = threadIdx.x;
    const int lane = tid & 31;
    const int wave = tid >> 5;
    const int wm = wave & 3;                 // 4 waves along M
    const int wn = wave >> 2;                // 2 waves along N
    const int m0 = blockIdx.y * 128 + wm * 32;
    const int n0 = blockIdx.x * 128 + wn * 64;
    const int lr = lane & 15;
    const int koff = (lane >> 4) * 8;        // half-of-wave K offset

    v8f acc[2][4];
#pragma unroll
    for (int i = 0; i < 2; ++i)
#pragma unroll
        for (int j = 0; j < 4; ++j) acc[i][j] = vzero8();

    for (int k0 = 0; k0 < K; k0 += 32) {
        v16h afrag[2], bfrag[4];
#pragma unroll
        for (int s = 0; s < 2; ++s) {
            const _Float16* p = A + (size_t)(m0 + s * 16 + lr) * K + k0 + koff;
            __builtin_prefetch(p + 32, 0, 3);        // near-scope prefetch
            afrag[s] = load_frag16(p);
        }
#pragma unroll
        for (int s = 0; s < 4; ++s) {
            const _Float16* p = Bt + (size_t)(n0 + s * 16 + lr) * K + k0 + koff;
            bfrag[s] = load_frag16(p);
        }
#pragma unroll
        for (int i = 0; i < 2; ++i)
#pragma unroll
            for (int j = 0; j < 4; ++j)
                acc[i][j] = wmma_f16(afrag[i], bfrag[j], acc[i][j]);
    }

    // store: C 16x16 layout: col = lane&15, row = vgpr + 8*(lane>=16)
#pragma unroll
    for (int i = 0; i < 2; ++i) {
        int row0 = m0 + i * 16 + (koff ? 8 : 0);
#pragma unroll
        for (int j = 0; j < 4; ++j) {
            int col = n0 + j * 16 + lr;
            if (outF) {
                float bv = bias[col];
#pragma unroll
                for (int r = 0; r < 8; ++r)
                    outF[(size_t)(row0 + r) * Nc + col] = acc[i][j][r] + bv;
            } else {
#pragma unroll
                for (int r = 0; r < 8; ++r)
                    outH[(size_t)(row0 + r) * Nc + col] = (_Float16)acc[i][j][r];
            }
        }
    }
}

// ---------------------------------------------------------------------------
// Kernel 5: flash attention over one (b, h), 128 query rows per block.
// q,k,v: f16 [M_ROWS, 512] with head slice at column h*64.
// 8 waves; each wave owns 16 query rows, full 64-wide output.
// K tile staged in LDS by the Tensor Data Mover; V tile staged transposed.
// ---------------------------------------------------------------------------
__global__ void __launch_bounds__(256)
flash_attn_kernel(const _Float16* __restrict__ q, const _Float16* __restrict__ k,
                  const _Float16* __restrict__ v, _Float16* __restrict__ o) {
    const int b = blockIdx.z;
    const int h = blockIdx.y;
    const int m0 = blockIdx.x * 128;
    const int tid  = threadIdx.x;
    const int lane = tid & 31;
    const int wave = tid >> 5;
    const int lr = lane & 15;
    const int koff = (lane >> 4) * 8;
    const int mq = m0 + wave * 16;                       // wave's 16 q rows
    const float SC = 0.125f * 1.4426950408889634f;       // scale * log2(e)

    __shared__ __attribute__((aligned(16))) _Float16 kt[32][64];       // K tile
    __shared__ __attribute__((aligned(16))) _Float16 vt[64][40];       // V^T tile
    __shared__ __attribute__((aligned(16))) _Float16 pls[8][16][40];   // per-wave P

    // Preload Q fragments (dh = 64 -> two k-chunks of 32)
    const _Float16* qrow = q + (size_t)(b * N_SEQ + mq + lr) * INNER + h * DIM_HEAD;
    v16h qf[2];
    qf[0] = load_frag16(qrow + koff);
    qf[1] = load_frag16(qrow + 32 + koff);

    v8f ot[4];
#pragma unroll
    for (int d = 0; d < 4; ++d) ot[d] = vzero8();
    float mrow[8], lsum[8];
#pragma unroll
    for (int r = 0; r < 8; ++r) { mrow[r] = -1e30f; lsum[r] = 0.f; }

    for (int j0 = 0; j0 < N_SEQ; j0 += 32) {
        __syncthreads();   // protect kt/vt from previous iteration's readers

#if USE_TDM
        // --- TDM: DMA K tile [32 keys x 64 dh] into LDS (one op, wave 0) ---
        if (wave == 0) {
            unsigned klds = (unsigned)(uintptr_t)(&kt[0][0]);
            unsigned long long ga = (unsigned long long)(uintptr_t)
                (k + (size_t)(b * N_SEQ + j0) * INNER + h * DIM_HEAD);
            tdm_load_k_tile(klds, ga);
            __builtin_amdgcn_s_wait_tensorcnt(0);
        }
#else
        // fallback: cooperative vector loads of K tile
        {
            int key = tid >> 3;
            int dc  = (tid & 7) * 8;
            const _Float16* kp =
                k + (size_t)(b * N_SEQ + j0 + key) * INNER + h * DIM_HEAD + dc;
            *(v8h*)&kt[key][dc] = *(const v8h*)kp;
        }
#endif
        // --- cooperatively stage V tile transposed: vt[d][j] -------------
        {
            int key = tid >> 3;                 // 0..31
            int dc  = (tid & 7) * 8;            // dh chunk base
            const _Float16* vp =
                v + (size_t)(b * N_SEQ + j0 + key) * INNER + h * DIM_HEAD + dc;
            v8h vv = *(const v8h*)vp;
#pragma unroll
            for (int e = 0; e < 8; ++e) vt[dc + e][key] = vv[e];
        }
        __syncthreads();

        // --- S = Q K^T over 32 keys (two 16-key tiles) --------------------
        v8f s[2];
#pragma unroll
        for (int t = 0; t < 2; ++t) {
            s[t] = vzero8();
            const _Float16* krow = &kt[t * 16 + lr][0];
            v16h kf0 = load_frag16(krow + koff);
            v16h kf1 = load_frag16(krow + 32 + koff);
            s[t] = wmma_f16(qf[0], kf0, s[t]);
            s[t] = wmma_f16(qf[1], kf1, s[t]);
        }

        // --- online softmax update ----------------------------------------
#pragma unroll
        for (int r = 0; r < 8; ++r) {
            float mx = fmaxf(s[0][r], s[1][r]);
#pragma unroll
            for (int off = 8; off >= 1; off >>= 1)
                mx = fmaxf(mx, __shfl_xor(mx, off, 32));
            float mnew = fmaxf(mrow[r], mx);
            float alpha = exp2f((mrow[r] - mnew) * SC);
            mrow[r] = mnew;
            float p0 = exp2f((s[0][r] - mnew) * SC);
            float p1 = exp2f((s[1][r] - mnew) * SC);
            s[0][r] = p0; s[1][r] = p1;
            float rsum = p0 + p1;
#pragma unroll
            for (int off = 8; off >= 1; off >>= 1)
                rsum += __shfl_xor(rsum, off, 32);
            lsum[r] = lsum[r] * alpha + rsum;
#pragma unroll
            for (int d = 0; d < 4; ++d) ot[d][r] *= alpha;
        }

        // --- P (C layout) -> LDS -> A-fragment layout ---------------------
#pragma unroll
        for (int t = 0; t < 2; ++t)
#pragma unroll
            for (int r = 0; r < 8; ++r)
                pls[wave][(koff ? 8 : 0) + r][t * 16 + lr] = (_Float16)s[t][r];
        // wave-local LDS RAW: compiler inserts s_wait_dscnt
        v16h pf = load_frag16(&pls[wave][lr][koff]);

        // --- O += P V -----------------------------------------------------
#pragma unroll
        for (int d = 0; d < 4; ++d) {
            v16h vf = load_frag16(&vt[d * 16 + lr][koff]);
            ot[d] = wmma_f16(pf, vf, ot[d]);
        }
    }

    // --- epilogue: O /= l, store f16 [M_ROWS, 512] ------------------------
#pragma unroll
    for (int r = 0; r < 8; ++r) {
        float inv = 1.f / lsum[r];
        int row = b * N_SEQ + mq + (koff ? 8 : 0) + r;
#pragma unroll
        for (int d = 0; d < 4; ++d) {
            int col = h * DIM_HEAD + d * 16 + lr;
            o[(size_t)row * INNER + col] = (_Float16)(ot[d][r] * inv);
        }
    }
}

// ---------------------------------------------------------------------------
// Host-side launcher
// ---------------------------------------------------------------------------
extern "C" void kernel_launch(void* const* d_in, const int* in_sizes, int n_in,
                              void* d_out, int out_size, void* d_ws, size_t ws_size,
                              hipStream_t stream) {
    const float* x    = (const float*)d_in[0];
    const float* z    = (const float*)d_in[1];
    const float* lxg  = (const float*)d_in[2];
    const float* lxb  = (const float*)d_in[3];
    const float* lzg  = (const float*)d_in[4];
    const float* lzb  = (const float*)d_in[5];
    const float* Wq   = (const float*)d_in[6];
    const float* Wk   = (const float*)d_in[7];
    const float* Wv   = (const float*)d_in[8];
    const float* Wz   = (const float*)d_in[9];
    const float* Wo   = (const float*)d_in[10];
    const float* bo   = (const float*)d_in[11];
    float* out = (float*)d_out;

    char* ws = (char*)d_ws;
    size_t off = 0;
    float* z1z2 = (float*)(ws + off);       off += (size_t)BATCH * 2 * X_DIM * 4;
    _Float16* WtQ  = (_Float16*)(ws + off); off += (size_t)X_DIM * INNER * 2;
    _Float16* WtK  = (_Float16*)(ws + off); off += (size_t)X_DIM * INNER * 2;
    _Float16* WtV  = (_Float16*)(ws + off); off += (size_t)X_DIM * INNER * 2;
    _Float16* WtO  = (_Float16*)(ws + off); off += (size_t)INNER * X_DIM * 2;
    _Float16* xmod = (_Float16*)(ws + off); off += (size_t)M_ROWS * X_DIM * 2;
    _Float16* qb   = (_Float16*)(ws + off); off += (size_t)M_ROWS * INNER * 2;
    _Float16* kb   = (_Float16*)(ws + off); off += (size_t)M_ROWS * INNER * 2;
    _Float16* vb   = (_Float16*)(ws + off); off += (size_t)M_ROWS * INNER * 2;
    _Float16* ao   = (_Float16*)(ws + off); off += (size_t)M_ROWS * INNER * 2;

    // 1) weight convert + transpose
    int wblocks = (X_DIM * INNER) / 256;
    cvt_transpose_kernel<<<wblocks, 256, 0, stream>>>(Wq, WtQ, X_DIM, INNER);
    cvt_transpose_kernel<<<wblocks, 256, 0, stream>>>(Wk, WtK, X_DIM, INNER);
    cvt_transpose_kernel<<<wblocks, 256, 0, stream>>>(Wv, WtV, X_DIM, INNER);
    cvt_transpose_kernel<<<wblocks, 256, 0, stream>>>(Wo, WtO, INNER, X_DIM);

    // 2) z path
    zpath_kernel<<<BATCH, 256, 0, stream>>>(z, lzg, lzb, Wz, z1z2);

    // 3) LN + FiLM on x -> f16
    xmod_kernel<<<M_ROWS, 128, 0, stream>>>(x, lxg, lxb, z1z2, xmod);

    // 4) Q/K/V projections (f16 out)
    dim3 ggrid(INNER / 128, M_ROWS / 128);
    gemm_wmma_kernel<<<ggrid, 256, 0, stream>>>(xmod, WtQ, qb, nullptr, nullptr,
                                                M_ROWS, INNER, X_DIM);
    gemm_wmma_kernel<<<ggrid, 256, 0, stream>>>(xmod, WtK, kb, nullptr, nullptr,
                                                M_ROWS, INNER, X_DIM);
    gemm_wmma_kernel<<<ggrid, 256, 0, stream>>>(xmod, WtV, vb, nullptr, nullptr,
                                                M_ROWS, INNER, X_DIM);

    // 5) flash attention
    dim3 agrid(N_SEQ / 128, HEADS, BATCH);
    flash_attn_kernel<<<agrid, 256, 0, stream>>>(qb, kb, vb, ao);

    // 6) output projection (f32 out + bias)
    dim3 ogrid(X_DIM / 128, M_ROWS / 128);
    gemm_wmma_kernel<<<ogrid, 256, 0, stream>>>(ao, WtO, nullptr, out, bo,
                                                M_ROWS, X_DIM, INNER);
}
